// UnifiedVADModel256ms_11381663334537
// MI455X (gfx1250) — matrix operand order repaired and checked by reading, not compiled, not run on hardware
//
#include <hip/hip_runtime.h>
#include <hip/hip_bf16.h>
#include <math.h>

typedef __bf16 bf16;
typedef bf16  v16bf __attribute__((ext_vector_type(16)));
typedef float v8f   __attribute__((ext_vector_type(8)));

#define BATCH 8192
#define NCHUNK (8*BATCH)     /* 65536 */
#define NFRAME (NCHUNK*4)    /* 262144 */

__device__ __forceinline__ v8f wmma_bf16(v16bf a, v16bf b, v8f c) {
  return __builtin_amdgcn_wmma_f32_16x16x32_bf16(false, a, false, b, (short)0, c, false, false);
}

__device__ __forceinline__ v16bf zero_frag() {
  v16bf a;
#pragma unroll
  for (int e = 0; e < 16; ++e) a[e] = (bf16)0.f;
  return a;
}

// A-frag: lane holds row m, K cols {half*8+0..7, half*8+16..23} of a 16x32 tile.
__device__ __forceinline__ v16bf load_a_frag(const bf16* row, int half) {
  v16bf a;
#pragma unroll
  for (int e = 0; e < 8; ++e) {
    a[e]     = row[half * 8 + e];
    a[e + 8] = row[half * 8 + 16 + e];
  }
  return a;
}

// B-frag: lane holds N col n0+(lane&15), K rows {half*8+0..7, half*8+16..23}.
__device__ __forceinline__ v16bf load_b_frag(const bf16* bbase, int lane, int ldb, int n0) {
  int half = lane >> 4;
  const bf16* p = bbase + n0 + (lane & 15);
  v16bf b;
#pragma unroll
  for (int e = 0; e < 8; ++e) {
    b[e]     = p[(size_t)(half * 8 + e) * ldb];
    b[e + 8] = p[(size_t)(half * 8 + 16 + e) * ldb];
  }
  return b;
}

__device__ __forceinline__ float sigf(float x) { return 1.f / (1.f + __expf(-x)); }

// Async global->LDS copy (CDNA5 ASYNCcnt path). 16B per lane per call.
__device__ __forceinline__ void async_ld16(unsigned lds_byte_off, const float* gptr) {
  unsigned long long ga = (unsigned long long)(uintptr_t)gptr;
  asm volatile("global_load_async_to_lds_b128 %0, %1, off"
               :: "v"(lds_byte_off), "v"(ga) : "memory");
}
__device__ __forceinline__ void wait_async0() {
  asm volatile("s_wait_asynccnt 0x0" ::: "memory");
}

// ------------------- weight prep -------------------

__global__ void k_prep_stft(const float* __restrict__ w, bf16* __restrict__ re,
                            bf16* __restrict__ im) {
  int idx = blockIdx.x * 256 + threadIdx.x;       // 256 K rows x 160 padded bins
  if (idx >= 256 * 160) return;
  int k = idx / 160, n = idx % 160;
  re[idx] = (n < 129) ? (bf16)w[(size_t)n * 256 + k]         : (bf16)0.f;
  im[idx] = (n < 129) ? (bf16)w[(size_t)(129 + n) * 256 + k] : (bf16)0.f;
}

__global__ void k_prep_conv(const float* __restrict__ w, bf16* __restrict__ Bm,
                            int Cin, int Cpad, int Cout, int taps, int tapOff, int ktaps) {
  int total = Cpad * taps * Cout;
  int idx = blockIdx.x * 256 + threadIdx.x;
  if (idx >= total) return;
  int k = idx / Cout, o = idx % Cout;
  int t = k / Cpad, c = k % Cpad;
  Bm[idx] = (c < Cin) ? (bf16)w[((size_t)o * Cin + c) * ktaps + t + tapOff] : (bf16)0.f;
}

__global__ void k_prep_lstm(const float* __restrict__ wih, const float* __restrict__ whh,
                            bf16* __restrict__ Bih, bf16* __restrict__ Bhh) {
  int idx = blockIdx.x * 256 + threadIdx.x;       // 128 K x 512 N
  if (idx >= 128 * 512) return;
  int k = idx / 512, n = idx % 512;
  Bih[idx] = (bf16)wih[(size_t)n * 128 + k];
  Bhh[idx] = (bf16)whh[(size_t)n * 128 + k];
}

// ------------------- STFT + magnitude -------------------
// 16 frames (= 4 chunks) per wave; GEMM [16x256] x [256x160] for re & im banks.
// Chunk audio staged to LDS via GLOBAL_LOAD_ASYNC_TO_LDS_B128 (ASYNCcnt).
__global__ void k_stft_mag(const float* __restrict__ audio, const bf16* __restrict__ wre,
                           const bf16* __restrict__ wim, bf16* __restrict__ mag) {
  __shared__ float lch[4 * 576];
  const int lane = threadIdx.x;
  const int g0 = blockIdx.x * 4;
  __builtin_prefetch(wre, 0, 0);
  // 4*576 floats = 576 x 16B chunks; rows contiguous & 16B aligned on both sides.
  for (int idx = lane; idx < 576; idx += 32) {
    int cl = idx / 144, qq = (idx % 144) * 4;
    int g = g0 + cl;
    int batch = g & (BATCH - 1);
    int ic = g >> 13;                         // chunk time index (B = 2^13)
    const float* gp = audio + (size_t)batch * 4160 + ic * 512 + qq;
    unsigned loff = (unsigned)(uintptr_t)(&lch[cl * 576 + qq]);
    async_ld16(loff, gp);
  }
  wait_async0();
  __syncthreads();
  const int half = lane >> 4;
  const int m = lane & 15;
  const int cl = m >> 2, j = m & 3;
  const v8f zac = {};
  for (int pass = 0; pass < 2; ++pass) {      // N tiles [0..5) and [5..10)
    v8f accR[5], accI[5];
#pragma unroll
    for (int q = 0; q < 5; ++q) { accR[q] = zac; accI[q] = zac; }
    for (int kt = 0; kt < 8; ++kt) {
      v16bf a;
#pragma unroll
      for (int e = 0; e < 16; ++e) {
        int K = kt * 32 + (e & 7) + ((e & 8) ? 16 : 0) + half * 8;
        int p = j * 128 + K;                  // padded window index [0,704)
        int q = (p < 64) ? (64 - p) : ((p < 640) ? (p - 64) : (1214 - p));  // reflect
        a[e] = (bf16)lch[cl * 576 + q];
      }
#pragma unroll
      for (int q = 0; q < 5; ++q) {
        int n0 = (pass * 5 + q) * 16;
        v16bf br = load_b_frag(wre + (size_t)(kt * 32) * 160, lane, 160, n0);
        accR[q] = wmma_bf16(a, br, accR[q]);
        v16bf bi = load_b_frag(wim + (size_t)(kt * 32) * 160, lane, 160, n0);
        accI[q] = wmma_bf16(a, bi, accI[q]);
      }
    }
#pragma unroll
    for (int q = 0; q < 5; ++q) {
      int n = (pass * 5 + q) * 16 + (lane & 15);
#pragma unroll
      for (int r = 0; r < 8; ++r) {
        int mm = r + half * 8;
        float re = accR[q][r], im = accI[q][r];
        mag[((size_t)blockIdx.x * 16 + mm) * 160 + n] = (bf16)sqrtf(re * re + im * im);
      }
    }
  }
}

// ------------------- encoder convs as implicit GEMMs -------------------

__global__ void k_conv1(const bf16* __restrict__ mag, const bf16* __restrict__ B1,
                        const float* __restrict__ b1, bf16* __restrict__ h1) {
  const int lane = threadIdx.x, half = lane >> 4, m = lane & 15;
  const int row = blockIdx.x * 16 + m;        // = g*4 + j
  const int j = row & 3;
  const v8f zac = {};
  v8f acc[8];
#pragma unroll
  for (int q = 0; q < 8; ++q) acc[q] = zac;
  for (int kt = 0; kt < 15; ++kt) {           // K = 3*160
    int t = kt / 5, cb = (kt % 5) * 32;
    int jj = j + t - 1;
    v16bf a = (jj >= 0 && jj < 4)
                  ? load_a_frag(mag + (size_t)(row + (jj - j)) * 160 + cb, half)
                  : zero_frag();
#pragma unroll
    for (int q = 0; q < 8; ++q) {
      v16bf b = load_b_frag(B1 + (size_t)(kt * 32) * 128, lane, 128, q * 16);
      acc[q] = wmma_bf16(a, b, acc[q]);
    }
  }
#pragma unroll
  for (int q = 0; q < 8; ++q) {
    int n = q * 16 + (lane & 15);
    float bias = b1[n];
#pragma unroll
    for (int r = 0; r < 8; ++r) {
      int mm = r + half * 8;
      h1[((size_t)(blockIdx.x * 16 + mm)) * 128 + n] = (bf16)fmaxf(acc[q][r] + bias, 0.f);
    }
  }
}

__global__ void k_conv2(const bf16* __restrict__ h1, const bf16* __restrict__ B2,
                        const float* __restrict__ b2, bf16* __restrict__ h2) {
  const int lane = threadIdx.x, half = lane >> 4, m = lane & 15;
  const int row = blockIdx.x * 16 + m;        // = g*2 + u
  const int g = row >> 1, u = row & 1;
  const v8f zac = {};
  v8f acc[4];
#pragma unroll
  for (int q = 0; q < 4; ++q) acc[q] = zac;
  for (int kt = 0; kt < 12; ++kt) {           // K = 3*128
    int t = kt >> 2, cb = (kt & 3) * 32;
    int jj = 2 * u - 1 + t;
    v16bf a = (jj >= 0 && jj < 4)
                  ? load_a_frag(h1 + (size_t)(g * 4 + jj) * 128 + cb, half)
                  : zero_frag();
#pragma unroll
    for (int q = 0; q < 4; ++q) {
      v16bf b = load_b_frag(B2 + (size_t)(kt * 32) * 64, lane, 64, q * 16);
      acc[q] = wmma_bf16(a, b, acc[q]);
    }
  }
#pragma unroll
  for (int q = 0; q < 4; ++q) {
    int n = q * 16 + (lane & 15);
    float bias = b2[n];
#pragma unroll
    for (int r = 0; r < 8; ++r) {
      int mm = r + half * 8;
      h2[((size_t)(blockIdx.x * 16 + mm)) * 64 + n] = (bf16)fmaxf(acc[q][r] + bias, 0.f);
    }
  }
}

__global__ void k_conv3(const bf16* __restrict__ h2, const bf16* __restrict__ B3,
                        const float* __restrict__ b3, bf16* __restrict__ h3) {
  const int lane = threadIdx.x, half = lane >> 4, m = lane & 15;
  const int g = blockIdx.x * 16 + m;
  const v8f zac = {};
  v8f acc[4];
#pragma unroll
  for (int q = 0; q < 4; ++q) acc[q] = zac;
#pragma unroll
  for (int kt = 0; kt < 4; ++kt) {            // K = 2*64 (taps 1,2 valid)
    int fr = kt >> 1, cb = (kt & 1) * 32;
    v16bf a = load_a_frag(h2 + (size_t)(g * 2 + fr) * 64 + cb, half);
#pragma unroll
    for (int q = 0; q < 4; ++q) {
      v16bf b = load_b_frag(B3 + (size_t)(kt * 32) * 64, lane, 64, q * 16);
      acc[q] = wmma_bf16(a, b, acc[q]);
    }
  }
#pragma unroll
  for (int q = 0; q < 4; ++q) {
    int n = q * 16 + (lane & 15);
    float bias = b3[n];
#pragma unroll
    for (int r = 0; r < 8; ++r) {
      int mm = r + half * 8;
      h3[((size_t)(blockIdx.x * 16 + mm)) * 64 + n] = (bf16)fmaxf(acc[q][r] + bias, 0.f);
    }
  }
}

__global__ void k_conv4(const bf16* __restrict__ h3, const bf16* __restrict__ B4,
                        const float* __restrict__ b4, bf16* __restrict__ feats) {
  const int lane = threadIdx.x, half = lane >> 4, m = lane & 15;
  const int g = blockIdx.x * 16 + m;
  const v8f zac = {};
  v8f acc[8];
#pragma unroll
  for (int q = 0; q < 8; ++q) acc[q] = zac;
#pragma unroll
  for (int kt = 0; kt < 2; ++kt) {            // K = 64 (center tap only)
    v16bf a = load_a_frag(h3 + (size_t)g * 64 + kt * 32, half);
#pragma unroll
    for (int q = 0; q < 8; ++q) {
      v16bf b = load_b_frag(B4 + (size_t)(kt * 32) * 128, lane, 128, q * 16);
      acc[q] = wmma_bf16(a, b, acc[q]);
    }
  }
#pragma unroll
  for (int q = 0; q < 8; ++q) {
    int n = q * 16 + (lane & 15);
    float bias = b4[n];
#pragma unroll
    for (int r = 0; r < 8; ++r) {
      int mm = r + half * 8;
      feats[((size_t)(blockIdx.x * 16 + mm)) * 128 + n] = (bf16)fmaxf(acc[q][r] + bias, 0.f);
    }
  }
}

// ------------------- LSTM (8 steps) + head + final -------------------
// One wave owns 16 batch rows across all 8 steps (no cross-block dependency).
__global__ void k_lstm(const bf16* __restrict__ feats, const bf16* __restrict__ Bih,
                       const bf16* __restrict__ Bhh, const float* __restrict__ b_ih,
                       const float* __restrict__ b_hh, const float* __restrict__ h0,
                       const float* __restrict__ c0, const float* __restrict__ out_w,
                       const float* __restrict__ out_b, float* __restrict__ out) {
  __shared__ float gates[16 * 512];
  __shared__ float cst[16 * 128];
  __shared__ float hfv[16 * 128];
  __shared__ bf16  hb[16 * 128];
  const int lane = threadIdx.x, half = lane >> 4, m = lane & 15;
  const int batch0 = blockIdx.x * 16;
  const v8f zac = {};
  for (int idx = lane; idx < 16 * 128; idx += 32) {
    int mm = idx >> 7, u = idx & 127;
    float hv = h0[(size_t)(batch0 + mm) * 128 + u];
    float cv = c0[(size_t)(batch0 + mm) * 128 + u];
    hfv[idx] = hv; hb[idx] = (bf16)hv; cst[idx] = cv;
  }
  float prodr = 1.f;
  __syncthreads();
  for (int st = 0; st < 8; ++st) {
    const bf16* xrow = feats + ((size_t)st * BATCH + batch0) * 128;
    for (int grp = 0; grp < 4; ++grp) {       // 4 groups x 8 N-tiles = 512 cols
      v8f acc[8];
#pragma unroll
      for (int q = 0; q < 8; ++q) acc[q] = zac;
#pragma unroll
      for (int kt = 0; kt < 4; ++kt) {
        v16bf ax = load_a_frag(xrow + (size_t)m * 128 + kt * 32, half);
        v16bf ah = load_a_frag(&hb[m * 128 + kt * 32], half);
#pragma unroll
        for (int q = 0; q < 8; ++q) {
          int n0 = (grp * 8 + q) * 16;
          v16bf bx = load_b_frag(Bih + (size_t)(kt * 32) * 512, lane, 512, n0);
          acc[q] = wmma_bf16(ax, bx, acc[q]);
          v16bf bh = load_b_frag(Bhh + (size_t)(kt * 32) * 512, lane, 512, n0);
          acc[q] = wmma_bf16(ah, bh, acc[q]);
        }
      }
#pragma unroll
      for (int q = 0; q < 8; ++q) {
        int n = (grp * 8 + q) * 16 + (lane & 15);
#pragma unroll
        for (int r = 0; r < 8; ++r) gates[(r + half * 8) * 512 + n] = acc[q][r];
      }
    }
    __syncthreads();
    for (int idx = lane; idx < 16 * 128; idx += 32) {
      int mm = idx >> 7, u = idx & 127;
      float ig = gates[mm * 512 + u]       + b_ih[u]       + b_hh[u];
      float fg = gates[mm * 512 + u + 128] + b_ih[u + 128] + b_hh[u + 128];
      float gg = gates[mm * 512 + u + 256] + b_ih[u + 256] + b_hh[u + 256];
      float og = gates[mm * 512 + u + 384] + b_ih[u + 384] + b_hh[u + 384];
      float cn = sigf(fg) * cst[idx] + sigf(ig) * tanhf(gg);
      float hn = sigf(og) * tanhf(cn);
      cst[idx] = cn; hfv[idx] = hn; hb[idx] = (bf16)hn;
    }
    __syncthreads();
    if (lane < 16) {
      float s = out_b[0];
      for (int u = 0; u < 128; ++u) s += hfv[lane * 128 + u] * out_w[u];
      prodr *= (1.f - sigf(s));
    }
    __syncthreads();
  }
  if (lane < 16) out[batch0 + lane] = 1.f - prodr;
  for (int idx = lane; idx < 16 * 128; idx += 32) {
    int mm = idx >> 7, u = idx & 127;
    out[(size_t)BATCH + (size_t)(batch0 + mm) * 128 + u] = hfv[idx];
    out[(size_t)BATCH + (size_t)BATCH * 128 + (size_t)(batch0 + mm) * 128 + u] = cst[idx];
  }
}

extern "C" void kernel_launch(void* const* d_in, const int* in_sizes, int n_in,
                              void* d_out, int out_size, void* d_ws, size_t ws_size,
                              hipStream_t stream) {
  (void)in_sizes; (void)n_in; (void)out_size; (void)ws_size;
  const float* audio = (const float*)d_in[0];
  const float* h0    = (const float*)d_in[1];
  const float* c0    = (const float*)d_in[2];
  const float* stftw = (const float*)d_in[3];
  const float* w1 = (const float*)d_in[4];  const float* b1 = (const float*)d_in[5];
  const float* w2 = (const float*)d_in[6];  const float* b2 = (const float*)d_in[7];
  const float* w3 = (const float*)d_in[8];  const float* b3 = (const float*)d_in[9];
  const float* w4 = (const float*)d_in[10]; const float* b4 = (const float*)d_in[11];
  const float* wih = (const float*)d_in[12]; const float* whh = (const float*)d_in[13];
  const float* bih = (const float*)d_in[14]; const float* bhh = (const float*)d_in[15];
  const float* outw = (const float*)d_in[16]; const float* outb = (const float*)d_in[17];
  float* out = (float*)d_out;

  char* ws = (char*)d_ws;
  size_t off = 0;
  auto alloc = [&](size_t bytes) -> void* {
    void* p = ws + off;
    off = (off + bytes + 255) & ~(size_t)255;
    return p;
  };
  bf16* wre = (bf16*)alloc(256 * 160 * sizeof(bf16));
  bf16* wim = (bf16*)alloc(256 * 160 * sizeof(bf16));
  bf16* B1  = (bf16*)alloc(480 * 128 * sizeof(bf16));
  bf16* B2  = (bf16*)alloc(384 * 64 * sizeof(bf16));
  bf16* B3  = (bf16*)alloc(128 * 64 * sizeof(bf16));
  bf16* B4  = (bf16*)alloc(64 * 128 * sizeof(bf16));
  bf16* Bih = (bf16*)alloc(128 * 512 * sizeof(bf16));
  bf16* Bhh = (bf16*)alloc(128 * 512 * sizeof(bf16));
  bf16* mag = (bf16*)alloc((size_t)NCHUNK * 4 * 160 * sizeof(bf16));
  bf16* h1  = (bf16*)alloc((size_t)NCHUNK * 4 * 128 * sizeof(bf16));
  bf16* h2  = (bf16*)alloc((size_t)NCHUNK * 2 * 64 * sizeof(bf16));
  bf16* h3  = (bf16*)alloc((size_t)NCHUNK * 64 * sizeof(bf16));
  bf16* feats = (bf16*)alloc((size_t)NCHUNK * 128 * sizeof(bf16));

  k_prep_stft<<<160, 256, 0, stream>>>(stftw, wre, wim);
  k_prep_conv<<<240, 256, 0, stream>>>(w1, B1, 129, 160, 128, 3, 0, 3);
  k_prep_conv<<<96, 256, 0, stream>>>(w2, B2, 128, 128, 64, 3, 0, 3);
  k_prep_conv<<<32, 256, 0, stream>>>(w3, B3, 64, 64, 64, 2, 1, 3);
  k_prep_conv<<<32, 256, 0, stream>>>(w4, B4, 64, 64, 128, 1, 1, 3);
  k_prep_lstm<<<256, 256, 0, stream>>>(wih, whh, Bih, Bhh);

  k_stft_mag<<<NFRAME / 16, 32, 0, stream>>>(audio, wre, wim, mag);
  k_conv1<<<NFRAME / 16, 32, 0, stream>>>(mag, B1, b1, h1);
  k_conv2<<<NCHUNK * 2 / 16, 32, 0, stream>>>(h1, B2, b2, h2);
  k_conv3<<<NCHUNK / 16, 32, 0, stream>>>(h2, B3, b3, h3);
  k_conv4<<<NCHUNK / 16, 32, 0, stream>>>(h3, B4, b4, feats);
  k_lstm<<<BATCH / 16, 32, 0, stream>>>(feats, Bih, Bhh, bih, bhh, h0, c0, outw, outb, out);
}
